// ForwardModelImages_71648644432349
// MI455X (gfx1250) — matrix operand order repaired and checked by reading, not compiled, not run on hardware
//
#include <hip/hip_runtime.h>

#define LRELU_SLOPE 0.01f
#define BN_EPS 1e-5f

typedef __attribute__((ext_vector_type(16))) _Float16 v16h;
typedef __attribute__((ext_vector_type(2)))  _Float16 h2;
typedef __attribute__((ext_vector_type(8)))  float    v8f;

// ---------------------------------------------------------------------------
// CNN encoder helpers (run once on a 128x128 image; perf-irrelevant)
// ---------------------------------------------------------------------------

__global__ void conv3x3s2_kernel(const float* __restrict__ in,
                                 const float* __restrict__ w,
                                 const float* __restrict__ b,
                                 float* __restrict__ out,
                                 int Cin, int Cout, int Hin, int Win,
                                 int Hout, int Wout) {
  int idx = blockIdx.x * blockDim.x + threadIdx.x;
  int total = Cout * Hout * Wout;
  if (idx >= total) return;
  int x  = idx % Wout;
  int y  = (idx / Wout) % Hout;
  int co = idx / (Wout * Hout);
  float acc = b[co];
  for (int ci = 0; ci < Cin; ++ci) {
    const float* ip = in + (size_t)ci * Hin * Win;
    const float* wp = w + (((size_t)co * Cin + ci) * 9);
#pragma unroll
    for (int ky = 0; ky < 3; ++ky)
#pragma unroll
      for (int kx = 0; kx < 3; ++kx)
        acc += ip[(2 * y + ky) * Win + (2 * x + kx)] * wp[ky * 3 + kx];
  }
  out[idx] = acc;
}

__global__ void bn_relu_kernel(float* __restrict__ x,
                               const float* __restrict__ g,
                               const float* __restrict__ beta,
                               int S) {
  int c = blockIdx.x;
  float* p = x + (size_t)c * S;
  __shared__ float s1[256];
  __shared__ float s2[256];
  float sum = 0.f, sq = 0.f;
  for (int i = threadIdx.x; i < S; i += blockDim.x) {
    float v = p[i];
    sum += v;
    sq += v * v;
  }
  s1[threadIdx.x] = sum;
  s2[threadIdx.x] = sq;
  __syncthreads();
  for (int off = 128; off > 0; off >>= 1) {
    if ((int)threadIdx.x < off) {
      s1[threadIdx.x] += s1[threadIdx.x + off];
      s2[threadIdx.x] += s2[threadIdx.x + off];
    }
    __syncthreads();
  }
  float m   = s1[0] / (float)S;
  float var = s2[0] / (float)S - m * m;
  float scale = g[c] * rsqrtf(var + BN_EPS);
  float shift = beta[c] - m * scale;
  for (int i = threadIdx.x; i < S; i += blockDim.x) {
    float v = p[i] * scale + shift;
    p[i] = v > 0.f ? v : 0.f;
  }
}

__global__ void prep_h0_kernel(const float* __restrict__ enc,   // 9
                               const float* __restrict__ ctrl,  // 4
                               const float* __restrict__ w0,    // 17x24
                               const float* __restrict__ b0,    // 24
                               float* __restrict__ h0pre) {     // 32 padded
  int m = threadIdx.x;
  if (m >= 32) return;
  float acc = 0.f;
  if (m < 24) {
    acc = b0[m];
#pragma unroll
    for (int j = 0; j < 4; ++j) acc += ctrl[j] * w0[(4 + j) * 24 + m];
#pragma unroll
    for (int j = 0; j < 9; ++j) acc += enc[j] * w0[(8 + j) * 24 + m];
  }
  h0pre[m] = acc;
}

// ---------------------------------------------------------------------------
// WMMA MLP over 1M particles (2 particle tiles of 16 per wave-iteration)
// ---------------------------------------------------------------------------

__device__ __forceinline__ v8f wmma_f16(v16h a, v16h b, v8f c) {
  return __builtin_amdgcn_wmma_f32_16x16x32_f16(false, a, false, b, (short)0, c,
                                                false, false);
}

union BFrag {
  unsigned u[8];
  v16h h;
};

__device__ __forceinline__ unsigned swap16(unsigned x) {
  return (unsigned)__shfl_xor((int)x, 16, 32);
}

__device__ __forceinline__ h2 pk16(float a, float b) {
  return __builtin_bit_cast(h2, __builtin_amdgcn_cvt_pkrtz(a, b));
}

// A fragment: W^T tile, W row-major [Kreal x Mdim].
// f16 A 16x32, contiguous k: lanes 0-15 -> M=lane, K in {0..15};
//                     lanes 16-31 -> M=lane-16, K in {16..31}.
__device__ v16h load_a_frag(const float* __restrict__ w, int Mdim, int Kreal,
                            int Mreal, int mt, int lane) {
  const int hi = (lane >= 16) ? 1 : 0;
  const int M  = mt * 16 + (lane & 15);
  v16h a;
#pragma unroll
  for (int v = 0; v < 16; ++v) {
    int K = v + 16 * hi;
    float val = (K < Kreal && M < Mreal) ? w[K * Mdim + M] : 0.f;
    a[v] = (_Float16)val;
  }
  return a;
}

// C fragment: broadcast bias[M] (C/D layout: lane=N, VGPR r -> M=mt*16+r+8*hi).
__device__ v8f load_c_frag(const float* __restrict__ bias, int len, int mt,
                           int lane) {
  const int hi = (lane >= 16) ? 1 : 0;
  v8f c;
#pragma unroll
  for (int r = 0; r < 8; ++r) {
    int M = mt * 16 + r + 8 * hi;
    c[r] = (M < len) ? bias[M] : 0.f;
  }
  return c;
}

// D tiles (pre-activation; features 0-15 in d0, 16-31 in d1) -> next-layer B
// fragment with leaky-ReLU fused in packed f16:
//   cvt_pk_rtz -> pk_mul(slope) -> pk_max  (lrelu(x) == max(x, slope*x))
// Cross-half exchange uses a pre-select so only 4 ds_bpermutes are needed:
// lanes<16 only consume the partner's p0, lanes>=16 only the partner's p1.
__device__ __forceinline__ v16h make_b_act(const v8f& d0, const v8f& d1,
                                           bool hi) {
  const h2 sl = {(_Float16)LRELU_SLOPE, (_Float16)LRELU_SLOPE};
  unsigned p0[4], p1[4];
#pragma unroll
  for (int v = 0; v < 4; ++v) {
    h2 a = pk16(d0[2 * v], d0[2 * v + 1]);
    h2 b = pk16(d1[2 * v], d1[2 * v + 1]);
    a = __builtin_elementwise_max(a, a * sl);
    b = __builtin_elementwise_max(b, b * sl);
    p0[v] = __builtin_bit_cast(unsigned, a);
    p1[v] = __builtin_bit_cast(unsigned, b);
  }
  BFrag bf;
#pragma unroll
  for (int v = 0; v < 4; ++v) {
    unsigned st = swap16(hi ? p0[v] : p1[v]);
    bf.u[v]     = hi ? st : p0[v];  // K = 2v,2v+1 (+16 on hi half)
    bf.u[v + 4] = hi ? p1[v] : st;  // K = 8+2v,.. (+16 on hi half)
  }
  return bf.h;
}

__global__ void __launch_bounds__(256) mlp_wmma_kernel(
    const float* __restrict__ ps,     // [N][4]
    const float* __restrict__ noise,  // [N][4]
    const float* __restrict__ h0pre,  // [32] padded
    const float* __restrict__ w0,     // [17][24]
    const float* __restrict__ w1,     // [24][24]
    const float* __restrict__ b1,     // [24]
    const float* __restrict__ w2,     // [24][8]
    const float* __restrict__ b2,     // [8]
    float* __restrict__ out,          // [N][4]
    int N) {
  const int lane = threadIdx.x & 31;
  const int wid  = (blockIdx.x * blockDim.x + threadIdx.x) >> 5;
  const int nw   = (gridDim.x * blockDim.x) >> 5;
  const bool hi  = lane >= 16;

  v16h a00 = load_a_frag(w0, 24, 4, 24, 0, lane);
  v16h a01 = load_a_frag(w0, 24, 4, 24, 1, lane);
  v16h a10 = load_a_frag(w1, 24, 24, 24, 0, lane);
  v16h a11 = load_a_frag(w1, 24, 24, 24, 1, lane);
  v16h a20 = load_a_frag(w2, 8, 24, 8, 0, lane);

  v8f c00 = load_c_frag(h0pre, 32, 0, lane);
  v8f c01 = load_c_frag(h0pre, 32, 1, lane);
  v8f c10 = load_c_frag(b1, 24, 0, lane);
  v8f c11 = load_c_frag(b1, 24, 1, lane);
  v8f c20 = load_c_frag(b2, 8, 0, lane);

  const int pairs = (N + 31) >> 5;  // 32 particles per iteration
  for (int t = wid; t < pairs; t += nw) {
    const int base = t * 32;
    const int pAll = base + lane;
    const int pcl  = pAll < N ? pAll : N - 1;

    // Full-wave coalesced loads: lane L owns particle base+L.
    float4 xs = *(const float4*)(ps + (size_t)pcl * 4);
    float4 nz = *(const float4*)(noise + (size_t)pcl * 4);

    int tn = t + nw;
    if (tn < pairs) {  // uniform; lowers to global_prefetch_b8
      __builtin_prefetch(ps + (size_t)tn * 128 + (size_t)lane * 4, 0, 1);
      __builtin_prefetch(noise + (size_t)tn * 128 + (size_t)lane * 4, 0, 1);
    }

    // Pack particle state, get the other tile's state via a half-swap.
    unsigned q0 = __builtin_bit_cast(unsigned, pk16(xs.x, xs.y));
    unsigned q1 = __builtin_bit_cast(unsigned, pk16(xs.z, xs.w));
    unsigned r0 = swap16(q0), r1 = swap16(q1);

    BFrag bA, bB;
#pragma unroll
    for (int v = 0; v < 8; ++v) { bA.u[v] = 0u; bB.u[v] = 0u; }
    bA.u[0] = hi ? 0u : q0;  // tile A: particles base+0..15
    bA.u[1] = hi ? 0u : q1;
    bB.u[0] = hi ? 0u : r0;  // tile B: particles base+16..31
    bB.u[1] = hi ? 0u : r1;

    // Layer 0 (K=4 padded to 32); two independent tiles fill hazard slots.
    v8f dA0 = wmma_f16(a00, bA.h, c00);
    v8f dB0 = wmma_f16(a00, bB.h, c00);
    v8f dA1 = wmma_f16(a01, bA.h, c01);
    v8f dB1 = wmma_f16(a01, bB.h, c01);

    v16h b1A = make_b_act(dA0, dA1, hi);
    v16h b1B = make_b_act(dB0, dB1, hi);
    v8f eA0 = wmma_f16(a10, b1A, c10);
    v8f eB0 = wmma_f16(a10, b1B, c10);
    v8f eA1 = wmma_f16(a11, b1A, c11);
    v8f eB1 = wmma_f16(a11, b1B, c11);

    v16h b2A = make_b_act(eA0, eA1, hi);
    v16h b2B = make_b_act(eB0, eB1, hi);
    v8f oA = wmma_f16(a20, b2A, c20);
    v8f oB = wmma_f16(a20, b2B, c20);

    // Tile B's noise lives in lanes 16-31; swap it down for the epilogue.
    float nzBx = __shfl_xor(nz.x, 16, 32);
    float nzBy = __shfl_xor(nz.y, 16, 32);
    float nzBz = __shfl_xor(nz.z, 16, 32);
    float nzBw = __shfl_xor(nz.w, 16, 32);

    // Lanes 0-15 hold out features 0..7 (mean 0-3, log-cov 4-7) per tile.
    if (!hi) {
      const int pA = base + lane;
      const int pB = pA + 16;
      if (pA < N) {
        float4 r;
        r.x = oA[0] + sqrtf(fminf(expf(oA[4]), 1.f)) * nz.x;
        r.y = oA[1] + sqrtf(fminf(expf(oA[5]), 1.f)) * nz.y;
        r.z = oA[2] + sqrtf(fminf(expf(oA[6]), 1.f)) * nz.z;
        r.w = oA[3] + sqrtf(fminf(expf(oA[7]), 1.f)) * nz.w;
        *(float4*)(out + (size_t)pA * 4) = r;
      }
      if (pB < N) {
        float4 r;
        r.x = oB[0] + sqrtf(fminf(expf(oB[4]), 1.f)) * nzBx;
        r.y = oB[1] + sqrtf(fminf(expf(oB[5]), 1.f)) * nzBy;
        r.z = oB[2] + sqrtf(fminf(expf(oB[6]), 1.f)) * nzBz;
        r.w = oB[3] + sqrtf(fminf(expf(oB[7]), 1.f)) * nzBw;
        *(float4*)(out + (size_t)pB * 4) = r;
      }
    }
  }
}

// ---------------------------------------------------------------------------
// Launch
// ---------------------------------------------------------------------------

static inline int cdiv(int a, int b) { return (a + b - 1) / b; }

extern "C" void kernel_launch(void* const* d_in, const int* in_sizes, int n_in,
                              void* d_out, int out_size, void* d_ws,
                              size_t ws_size, hipStream_t stream) {
  const float* ps    = (const float*)d_in[0];   // (1,N,4)
  const float* ctrl  = (const float*)d_in[1];   // (1,1,4)
  const float* image = (const float*)d_in[2];   // (1,1,128,128)
  const float* noise = (const float*)d_in[3];   // (N,4)
  const float* cw[5], *cb[5], *bg[5], *bb[5];
  for (int i = 0; i < 5; ++i) {
    cw[i] = (const float*)d_in[4 + 4 * i + 0];
    cb[i] = (const float*)d_in[4 + 4 * i + 1];
    bg[i] = (const float*)d_in[4 + 4 * i + 2];
    bb[i] = (const float*)d_in[4 + 4 * i + 3];
  }
  const float* w0 = (const float*)d_in[24];
  const float* b0 = (const float*)d_in[25];
  const float* w1 = (const float*)d_in[26];
  const float* b1 = (const float*)d_in[27];
  const float* w2 = (const float*)d_in[28];
  const float* b2 = (const float*)d_in[29];
  float* out = (float*)d_out;
  const int N = in_sizes[0] / 4;

  // Workspace (floats): act0 [0,40000), act1 [40000,50000), h0pre [50000,50032)
  float* act0  = (float*)d_ws;
  float* act1  = act0 + 40000;
  float* h0pre = act0 + 50000;

  const int Hs[6] = {128, 63, 31, 15, 7, 3};
  const int Cs[6] = {1, 8, 8, 8, 8, 1};
  const float* src = image;
  float* bufs[2] = {act0, act1};
  for (int l = 0; l < 5; ++l) {
    float* dst = bufs[l & 1];
    int Hin = Hs[l], Hout = Hs[l + 1];
    int Cin = Cs[l], Cout = Cs[l + 1];
    int total = Cout * Hout * Hout;
    conv3x3s2_kernel<<<cdiv(total, 256), 256, 0, stream>>>(
        src, cw[l], cb[l], dst, Cin, Cout, Hin, Hin, Hout, Hout);
    bn_relu_kernel<<<Cout, 256, 0, stream>>>(dst, bg[l], bb[l], Hout * Hout);
    src = dst;
  }
  prep_h0_kernel<<<1, 32, 0, stream>>>(src, ctrl, w0, b0, h0pre);

  mlp_wmma_kernel<<<512, 256, 0, stream>>>(ps, noise, h0pre, w0, w1, b1, w2,
                                           b2, out, N);
}
